// LGL_4612794876091
// MI455X (gfx1250) — matrix operand order repaired and checked by reading, not compiled, not run on hardware
//
#include <hip/hip_runtime.h>

// CDNA5 / gfx1250, wave32. Matrix ops via v_wmma_f32_16x16x32_f16.
typedef __attribute__((ext_vector_type(16))) _Float16 v16h;
typedef __attribute__((ext_vector_type(8)))  float    v8f;

#define NND   64      // nodes
#define KNB   32      // neighbors per node
#define FEAT  1024    // layer-1 feature dim
#define C1    4
#define F1    16
#define OUT1  64      // C1*F1
#define C2    32
#define NCLS  10
#define BN_EPS 1e-5f

union V16H { v16h v; _Float16 h[16]; };

// adj(a,b) = sgnroot(x_a*s_b + x_b*s_a) row-normalized over the (single) channel:
//          = sign(t) * q / (q + 1e-7),  q = sqrt(max(|t|, 1e-8)).
// Raw v_sqrt_f32 / v_rcp_f32 (result is rounded to f16 right after, so 1-ulp
// f32 accuracy is irrelevant). The fmin clamp keeps r finite even for t=+-Inf
// (unreachable for finite inputs; reference maps that case to 0 via a NaN
// check, which costs a serializing v_cmp->SGPR + v_cndmask per element).
__device__ __forceinline__ float adjval(float xa, float sa, float xb, float sb) {
    float t = fmaf(xa, sb, xb * sa);
    float q = __builtin_amdgcn_sqrtf(fmaxf(fabsf(t), 1e-8f));
    q = fminf(q, 1e18f);
    float r = q * __builtin_amdgcn_rcpf(q + 1e-7f);
    return copysignf(r, t);
}

// ---------------------------------------------------------------------------
// Kernel A: one block (8 waves) per node.
//  Phase 0: stage x, s = sum_k nb, f16 neighbor panel, f16 W1 panel in LDS.
//  Phase 1: fused adjacency GEMM  D[a, k] = sum_b adj(a,b) * nb[k][b]
//           (M=1024, N=32, K=1024) on v_wmma_f32_16x16x32_f16, with the f16
//           adjacency A-tiles synthesized in VGPRs; 2 a-tiles share each
//           B-tile load. xa[a] = sum_b adj(a,b) x[b] is accumulated on VALU
//           from the same f32 adjacency values (one column doesn't deserve a
//           WMMA) and reduced with a cross-half shuffle.
//  Phase 2: [na | xa] x W1^T (M=48 padded, N=64, K=1024) via WMMA from LDS.
// ---------------------------------------------------------------------------
extern "C" __global__ __launch_bounds__(256)
void lgl_layer1(const float* __restrict__ x, const float* __restrict__ nb,
                const float* __restrict__ W1,
                float* __restrict__ x2_pre,     // (64, 64)
                float* __restrict__ nb2_pre) {  // (64, 32, 64)
    extern __shared__ char smem[];
    float*     sX  = (float*)smem;                 // 1024 f32   (4 KB)
    float*     sS  = sX + FEAT;                    // 1024 f32   (4 KB)
    _Float16*  sNB = (_Float16*)(sS + FEAT);       // 32*1024 f16 (64 KB)
    _Float16*  sNA = sNB + KNB * FEAT;             // 48*1024 f16 (96 KB)
    _Float16*  sW1 = sNA + 48 * FEAT;              // 64*1024 f16 (128 KB)  -> 296 KB total

    const int node = blockIdx.x;
    const int tid  = threadIdx.x;
    const int wave = tid >> 5;
    const int lane = tid & 31;
    const int lrow = lane & 15;   // row/col within 16-wide tile
    const int lhi  = lane >> 4;   // upper half of wave32

    // ---- Phase 0: stage operands ------------------------------------------
    for (int b = tid; b < FEAT; b += 256) {
        sX[b] = x[node * FEAT + b];
        float sum = 0.f;
        const float* nbp = nb + (size_t)node * KNB * FEAT + b;
        #pragma unroll 4
        for (int k = 0; k < KNB; ++k) {
            float v = nbp[k * FEAT];           // coalesced across threads in b
            sNB[k * FEAT + b] = (_Float16)v;
            sum += v;
        }
        sS[b] = sum;
    }
    for (int i = tid; i < OUT1 * FEAT; i += 256)   // W1 f32 -> f16, coalesced
        sW1[i] = (_Float16)W1[i];
    __syncthreads();

    // ---- Phase 1 ----------------------------------------------------------
    // A operand (16x32 f16) VGPR v, lane L: row = L%16,
    //   K pair = (v&3)*2 + (L>=16 ? 8 : 0) + (v>=4 ? 16 : 0)   (ISA 7.12.2)
    // B operand (32x16 f16) VGPR v, lane L: col = L%16,
    //   K pair = (v&3)*2 + (L>=16 ? 8 : 0) + (v>=4 ? 16 : 0)
    for (int atp = wave; atp < 32; atp += 8) {     // pair of a-tiles per iter
        const int   a0 = atp * 32 + lrow;
        const int   a1 = a0 + 16;
        const float xr0 = sX[a0], sr0 = sS[a0];
        const float xr1 = sX[a1], sr1 = sS[a1];
        v8f c00 = {}, c01 = {}, c10 = {}, c11 = {};
        float xa0 = 0.f, xa1 = 0.f;
        for (int kk = 0; kk < FEAT / 32; ++kk) {
            const int k0 = kk * 32;
            V16H A0, A1;
            #pragma unroll
            for (int v = 0; v < 8; ++v) {
                const int kb = k0 + ((v & 3) * 2) + lhi * 8 + (v >> 2) * 16;
                const float xb0 = sX[kb],     sb0 = sS[kb];
                const float xb1 = sX[kb + 1], sb1 = sS[kb + 1];
                const float e00 = adjval(xr0, sr0, xb0, sb0);
                const float e01 = adjval(xr0, sr0, xb1, sb1);
                const float e10 = adjval(xr1, sr1, xb0, sb0);
                const float e11 = adjval(xr1, sr1, xb1, sb1);
                A0.h[2*v] = (_Float16)e00;  A0.h[2*v+1] = (_Float16)e01;
                A1.h[2*v] = (_Float16)e10;  A1.h[2*v+1] = (_Float16)e11;
                xa0 = fmaf(e00, xb0, fmaf(e01, xb1, xa0));   // xa column on VALU
                xa1 = fmaf(e10, xb0, fmaf(e11, xb1, xa1));
            }
            V16H B0, B1;
            #pragma unroll
            for (int v = 0; v < 8; ++v) {
                const int kb = k0 + ((v & 3) * 2) + lhi * 8 + (v >> 2) * 16;
                B0.h[2*v]   = sNB[lrow * FEAT + kb];          // neighbors 0..15
                B0.h[2*v+1] = sNB[lrow * FEAT + kb + 1];
                B1.h[2*v]   = sNB[(16 + lrow) * FEAT + kb];   // neighbors 16..31
                B1.h[2*v+1] = sNB[(16 + lrow) * FEAT + kb + 1];
            }
            c00 = __builtin_amdgcn_wmma_f32_16x16x32_f16(false, A0.v, false, B0.v, (short)0, c00, false, false);
            c01 = __builtin_amdgcn_wmma_f32_16x16x32_f16(false, A0.v, false, B1.v, (short)0, c01, false, false);
            c10 = __builtin_amdgcn_wmma_f32_16x16x32_f16(false, A1.v, false, B0.v, (short)0, c10, false, false);
            c11 = __builtin_amdgcn_wmma_f32_16x16x32_f16(false, A1.v, false, B1.v, (short)0, c11, false, false);
        }
        // D layout: VGPR r, lane L -> (row = r + 8*(L/16), col = L%16).
        // Store transposed into sNA[j][a] (j = GEMM2 row).
        #pragma unroll
        for (int r = 0; r < 8; ++r) {
            const int aa0 = atp * 32 + r + 8 * lhi;
            const int aa1 = aa0 + 16;
            sNA[lrow        * FEAT + aa0] = (_Float16)c00[r];
            sNA[(16 + lrow) * FEAT + aa0] = (_Float16)c01[r];
            sNA[lrow        * FEAT + aa1] = (_Float16)c10[r];
            sNA[(16 + lrow) * FEAT + aa1] = (_Float16)c11[r];
        }
        // lanes L and L+16 hold complementary K-halves of the same row
        xa0 += __shfl_xor(xa0, 16);
        xa1 += __shfl_xor(xa1, 16);
        if (lhi == 0) {
            sNA[KNB * FEAT + atp * 32 + lrow]      = (_Float16)xa0;
            sNA[KNB * FEAT + atp * 32 + 16 + lrow] = (_Float16)xa1;
        }
    }
    __syncthreads();

    // ---- Phase 2: [na | xa] x W1^T -> pre-BN layer-1 outputs --------------
    // sNA rows 33..47 are garbage; their output rows are discarded (D rows
    // depend only on their own A row).
    for (int p = wave; p < 12; p += 8) {          // 3 M-tiles x 4 N-tiles
        const int mt = p >> 2, nt = p & 3;
        v8f acc = {};
        for (int kk = 0; kk < FEAT / 32; ++kk) {
            const int k0 = kk * 32;
            V16H A, B;
            #pragma unroll
            for (int v = 0; v < 8; ++v) {
                const int ka = k0 + ((v & 3) * 2) + lhi * 8 + (v >> 2) * 16;
                A.h[2*v]   = sNA[(mt * 16 + lrow) * FEAT + ka];
                A.h[2*v+1] = sNA[(mt * 16 + lrow) * FEAT + ka + 1];
                const int kb = k0 + ((v & 3) * 2) + lhi * 8 + (v >> 2) * 16;
                B.h[2*v]   = sW1[(nt * 16 + lrow) * FEAT + kb];
                B.h[2*v+1] = sW1[(nt * 16 + lrow) * FEAT + kb + 1];
            }
            acc = __builtin_amdgcn_wmma_f32_16x16x32_f16(false, A.v, false, B.v, (short)0, acc, false, false);
        }
        const int o = nt * 16 + lrow;
        #pragma unroll
        for (int r = 0; r < 8; ++r) {
            const int m = mt * 16 + r + 8 * lhi;
            if (m < KNB)       nb2_pre[((size_t)node * KNB + m) * OUT1 + o] = acc[r];
            else if (m == KNB) x2_pre[node * OUT1 + o] = acc[r];
        }
    }
}

// ---------------------------------------------------------------------------
// Tail: all the tiny cross-node work in one block — BN1 (x over batch,
// neighbors per-node) + softsign, layer-2 4x16x16 adjacencies + contraction
// with W2, BN2 + softsign, final linear. ~60 KB static LDS, scalar VALU.
// ---------------------------------------------------------------------------
extern "C" __global__ __launch_bounds__(256)
void lgl_tail(const float* __restrict__ x2_pre, const float* __restrict__ nb2_pre,
              const float* __restrict__ bn1_w, const float* __restrict__ bn1_b,
              const float* __restrict__ bn2_w, const float* __restrict__ bn2_b,
              const float* __restrict__ W2, const float* __restrict__ lin_w,
              const float* __restrict__ lin_b, float* __restrict__ out) {
    __shared__ float sX1[NND * OUT1];
    __shared__ float sSNB[NND * OUT1];
    __shared__ float sXA[NND * OUT1];
    __shared__ float sY[NND * C2];
    __shared__ float sRed[256], sRed2[256];
    __shared__ float sM1[C1], sI1[C1];
    __shared__ float sM2[C2], sI2[C2];

    const int tid = threadIdx.x;

    // BN1 stats for x-path: per channel over (node, feature)
    {
        const int c = tid >> 6, n = tid & 63;
        float s = 0.f, s2 = 0.f;
        for (int f = 0; f < F1; ++f) {
            float v = x2_pre[n * OUT1 + c * F1 + f];
            s += v; s2 += v * v;
        }
        sRed[tid] = s; sRed2[tid] = s2;
    }
    __syncthreads();
    if (tid < C1) {
        float s = 0.f, s2 = 0.f;
        for (int n = 0; n < 64; ++n) { s += sRed[tid * 64 + n]; s2 += sRed2[tid * 64 + n]; }
        float m = s / 1024.f;
        sM1[tid] = m;
        sI1[tid] = rsqrtf(s2 / 1024.f - m * m + BN_EPS);
    }
    __syncthreads();
    // x1 = softsign(bn1(x2_pre))
    for (int i = tid; i < NND * OUT1; i += 256) {
        const int c = (i & 63) >> 4;
        float y = (x2_pre[i] - sM1[c]) * sI1[c] * bn1_w[c] + bn1_b[c];
        sX1[i] = y / (1.f + fabsf(y));
    }
    // Per-node neighbor BN (+softsign) folded directly into the K-sum snb.
    {
        const int n = tid >> 2, c = tid & 3;
        const float* base = nb2_pre + (size_t)n * KNB * OUT1 + c * F1;
        float s = 0.f, s2 = 0.f;
        for (int k = 0; k < KNB; ++k)
            for (int f = 0; f < F1; ++f) {
                float v = base[k * OUT1 + f];
                s += v; s2 += v * v;
            }
        float m   = s / 512.f;
        float inv = rsqrtf(s2 / 512.f - m * m + BN_EPS);
        float w = bn1_w[c], bb = bn1_b[c];
        float acc[F1];
        #pragma unroll
        for (int f = 0; f < F1; ++f) acc[f] = 0.f;
        for (int k = 0; k < KNB; ++k) {
            #pragma unroll
            for (int f = 0; f < F1; ++f) {
                float y = (base[k * OUT1 + f] - m) * inv * w + bb;
                acc[f] += y / (1.f + fabsf(y));
            }
        }
        for (int f = 0; f < F1; ++f) sSNB[n * OUT1 + c * F1 + f] = acc[f];
    }
    __syncthreads();
    // Layer-2 adjacency (channel-normalized) + xa2 = adj @ x1, unit = (n, a)
    for (int u = tid; u < NND * F1; u += 256) {
        const int n = u >> 4, a = u & 15;
        const float* x1n  = sX1  + n * OUT1;
        const float* snbn = sSNB + n * OUT1;
        float acc[C1] = {0.f, 0.f, 0.f, 0.f};
        for (int b = 0; b < F1; ++b) {
            float sc[C1], den = 1e-7f;
            #pragma unroll
            for (int c = 0; c < C1; ++c) {
                float t = x1n[c * F1 + a] * snbn[c * F1 + b]
                        + x1n[c * F1 + b] * snbn[c * F1 + a];
                float q = __builtin_amdgcn_sqrtf(fmaxf(fabsf(t), 1e-8f));
                sc[c] = copysignf(q, t);
                den  += q;
            }
            const float rden = __builtin_amdgcn_rcpf(den);
            #pragma unroll
            for (int c = 0; c < C1; ++c) {
                float adj = sc[c] * rden;
                adj = (adj != adj) ? 0.f : adj;
                acc[c] += adj * x1n[c * F1 + b];
            }
        }
        #pragma unroll
        for (int c = 0; c < C1; ++c) sXA[n * OUT1 + c * F1 + a] = acc[c];
    }
    __syncthreads();
    // x2 = W2 . xa2   (64 nodes x 32 outputs, K = 64)
    for (int u = tid; u < NND * C2; u += 256) {
        const int n = u >> 5, o = u & 31;
        const float* w  = W2  + o * OUT1;
        const float* xv = sXA + n * OUT1;
        float acc = 0.f;
        #pragma unroll 8
        for (int i = 0; i < OUT1; ++i) acc += w[i] * xv[i];
        sY[n * C2 + o] = acc;
    }
    __syncthreads();
    if (tid < C2) {
        float s = 0.f, s2 = 0.f;
        for (int n = 0; n < NND; ++n) { float v = sY[n * C2 + tid]; s += v; s2 += v * v; }
        float m = s / 64.f;
        sM2[tid] = m;
        sI2[tid] = rsqrtf(s2 / 64.f - m * m + BN_EPS);
    }
    __syncthreads();
    for (int i = tid; i < NND * C2; i += 256) {
        const int o = i & 31;
        float y = (sY[i] - sM2[o]) * sI2[o] * bn2_w[o] + bn2_b[o];
        sY[i] = y / (1.f + fabsf(y));
    }
    __syncthreads();
    for (int u = tid; u < NND * NCLS; u += 256) {
        const int n = u / NCLS, cls = u % NCLS;
        float acc = lin_b[cls];
        #pragma unroll
        for (int o = 0; o < C2; ++o) acc += sY[n * C2 + o] * lin_w[cls * C2 + o];
        out[u] = acc;
    }
}

extern "C" void kernel_launch(void* const* d_in, const int* in_sizes, int n_in,
                              void* d_out, int out_size, void* d_ws, size_t ws_size,
                              hipStream_t stream) {
    (void)in_sizes; (void)n_in; (void)out_size; (void)ws_size;
    const float* x    = (const float*)d_in[0];
    const float* nb   = (const float*)d_in[1];
    const float* W1   = (const float*)d_in[2];
    const float* W2   = (const float*)d_in[3];
    const float* bn1w = (const float*)d_in[4];
    const float* bn1b = (const float*)d_in[5];
    const float* bn2w = (const float*)d_in[6];
    const float* bn2b = (const float*)d_in[7];
    const float* lw   = (const float*)d_in[8];
    const float* lb   = (const float*)d_in[9];
    float* out = (float*)d_out;

    float* x2_pre  = (float*)d_ws;            // 64*64 f32
    float* nb2_pre = x2_pre + NND * OUT1;     // 64*32*64 f32  (total ~528 KB)

    // 4+4 KB x/s + 64 KB f16 neighbor panel + 96 KB f16 na panel
    // + 128 KB f16 W1 panel = 296 KB of the 320 KB CDNA5 WGP LDS
    // (occupancy was already 1 block/WGP at 168 KB, so this is free).
    const int lds = (int)((2 * FEAT) * sizeof(float)
                        + (KNB * FEAT + 48 * FEAT + OUT1 * FEAT) * sizeof(_Float16));
    hipFuncSetAttribute((const void*)lgl_layer1,
                        hipFuncAttributeMaxDynamicSharedMemorySize, lds);

    lgl_layer1<<<NND, 256, lds, stream>>>(x, nb, W1, x2_pre, nb2_pre);
    lgl_tail<<<1, 256, 0, stream>>>(x2_pre, nb2_pre, bn1w, bn1b, bn2w, bn2b,
                                    W2, lw, lb, out);
}
